// rollout_GNN_GRU_16707422781859
// MI455X (gfx1250) — matrix operand order repaired and checked by reading, not compiled, not run on hardware
//
#include <hip/hip_runtime.h>
#include <math.h>

// Problem constants (from the reference)
#define NN   20000
#define EE   320000
#define IND  64
#define HID  256
#define GH   256
#define TT   64

// ---------------- WMMA types ----------------
typedef __attribute__((ext_vector_type(16))) __bf16 v16bf;
typedef __attribute__((ext_vector_type(8)))  float  v8f;

union BF16x16 {
  v16bf v;
  unsigned short u[16];
};
static_assert(sizeof(BF16x16) == 32, "BF16x16 must be 32 bytes");

__device__ __forceinline__ unsigned short f32_to_bf16(float f) {
  union { float f; unsigned int u; } x;
  x.f = f;
  unsigned int u = x.u;
  u += 0x7fffu + ((u >> 16) & 1u);   // round-to-nearest-even
  return (unsigned short)(u >> 16);
}

__device__ __forceinline__ float sigmoidf_(float x) {
  return 1.0f / (1.0f + expf(-x));
}

// Flat index of element (row, k) inside the bf16 A-fragment-packed buffer.
// Fragment layout (per ISA 16-bit A 16x32): flat = ((mt*KT + kt)*32 + lane)*16 + i
//   lane<16 : row = mt*16+lane,    i 0..7 -> K 0..7,  i 8..15 -> K 16..23
//   lane>=16: row = mt*16+lane-16, i 0..7 -> K 8..15, i 8..15 -> K 24..31
__device__ __forceinline__ long long frag_index(int row, int k, int K) {
  int KT = K >> 5;
  int mt = row >> 4, rl = row & 15;
  int kt = k >> 5,  ko = k & 31;
  int lane, i;
  if (ko < 8)       { lane = rl;      i = ko;      }
  else if (ko < 16) { lane = rl + 16; i = ko - 8;  }
  else if (ko < 24) { lane = rl;      i = ko - 8;  }
  else              { lane = rl + 16; i = ko - 16; }
  return ((long long)(mt * KT + kt) * 32 + lane) * 16 + i;
}

// ---------------- Utility kernels ----------------
__global__ void zero_f32(float* __restrict__ p, int n) {
  int i = blockIdx.x * blockDim.x + threadIdx.x;
  if (i < n) p[i] = 0.0f;
}

__global__ void deg_accum(const int* __restrict__ dst, float* __restrict__ deg, int e) {
  int i = blockIdx.x * blockDim.x + threadIdx.x;
  if (i < e) atomicAdd(&deg[dst[i]], 1.0f);
}

__global__ void norm_from_deg(const float* __restrict__ deg, float* __restrict__ norm, int n) {
  int i = blockIdx.x * blockDim.x + threadIdx.x;
  if (i < n) {
    float d = deg[i];
    norm[i] = rsqrtf(d > 1.0f ? d : 1.0f);
  }
}

// temp = x0 (f32) and tempP = bf16 A-fragments of x0
__global__ void init_temp(const float* __restrict__ x0, float* __restrict__ temp,
                          unsigned short* __restrict__ tempP, int total) {
  int idx = blockIdx.x * blockDim.x + threadIdx.x;
  if (idx >= total) return;
  int row = idx / IND, k = idx - row * IND;
  float v = x0[idx];
  temp[idx] = v;
  tempP[frag_index(row, k, IND)] = f32_to_bf16(v);
}

// Pack weight matrix into per-lane B fragments for v_wmma_f32_16x16x32_bf16.
// B is logically [K x Nout]; transposed=1 means W is stored [Nout x K] (e.g. W_ih).
// Fragment layout: flat = ((nt*KT + kt)*32 + lane)*16 + i
//   lane<16 : col n = nt*16+lane,     i 0..7 -> k 0..7,  i 8..15 -> k 16..23
//   lane>=16: col n = nt*16+lane-16,  i 0..7 -> k 8..15, i 8..15 -> k 24..31
__global__ void pack_b(const float* __restrict__ W, unsigned short* __restrict__ Bp,
                       int K, int Nout, int transposed) {
  long long idx = (long long)blockIdx.x * blockDim.x + threadIdx.x;
  long long total = (long long)K * Nout;
  if (idx >= total) return;
  int i = (int)(idx & 15);
  long long r = idx >> 4;
  int lane = (int)(r & 31);
  r >>= 5;
  int KT = K >> 5;
  int kt = (int)(r % KT);
  int nt = (int)(r / KT);
  int n = nt * 16 + (lane & 15);
  int k = kt * 32 + ((lane < 16) ? 0 : 8) + ((i < 8) ? i : 8 + i);
  float val = transposed ? W[(long long)n * K + k] : W[(long long)k * Nout + n];
  Bp[idx] = f32_to_bf16(val);
}

// ---------------- WMMA GEMM ------------------------------------------------
// C[M x Nout] = unpack(Ap) * unpack(Bp) [+ bias]
// grid = (Nout/64, ceil((M/32)/8)); block = 256 threads = 8 waves.
// Each wave owns TWO M tiles (a pair) and the block's 4 N tiles: 8 WMMAs/K-tile.
// B panel (4 N tiles, all K) is staged in LDS once and shared by all 8 waves.
__global__ __launch_bounds__(256)
void wmma_gemm_packed(const unsigned short* __restrict__ Ap,
                      const unsigned short* __restrict__ Bp,
                      float* __restrict__ C, const float* __restrict__ bias,
                      int M, int K, int Nout) {
  const int KT = K >> 5;                 // <= 8
  const int lane = threadIdx.x & 31;
  const int wave = threadIdx.x >> 5;
  const int ntg = blockIdx.x;            // group of 4 consecutive N tiles
  const int mp  = blockIdx.y * 8 + wave; // M tile PAIR for this wave
  const int mt0 = mp * 2;

  __shared__ unsigned short Bsh[4 * 8 * 32 * 16];   // 32 KB max (KT=8)

  // Cooperative stage of the 4 N tiles' B fragments into LDS (contiguous range).
  {
    const uint4* bg = (const uint4*)(Bp + (long long)ntg * 4 * KT * 32 * 16);
    uint4* bs = (uint4*)Bsh;
    const int nvec = 4 * KT * 32 * 16 / 8;          // uint4 elements
    for (int v = threadIdx.x; v < nvec; v += 256) bs[v] = bg[v];
  }
  __syncthreads();
  if (mt0 * 16 >= M) return;             // wave-uniform (M % 32 == 0): EXEC all-ones

  const unsigned short* abase0 = Ap + ((long long)mt0 * KT * 32 + lane) * 16;
  const unsigned short* abase1 = abase0 + (long long)KT * 512;   // mt0+1

  v8f acc[2][4];
#pragma unroll
  for (int mi = 0; mi < 2; ++mi)
#pragma unroll
    for (int j = 0; j < 4; ++j) acc[mi][j] = (v8f){0.f,0.f,0.f,0.f,0.f,0.f,0.f,0.f};

  for (int kt = 0; kt < KT; ++kt) {
    const unsigned short* ap0 = abase0 + (long long)kt * 512;
    const unsigned short* ap1 = abase1 + (long long)kt * 512;
    if (kt + 1 < KT) __builtin_prefetch(ap0 + 512, 0, 3);   // global_prefetch_b8
    BF16x16 a0 = *(const BF16x16*)ap0;
    BF16x16 a1 = *(const BF16x16*)ap1;

    // Load all 4 B fragments into distinct registers first (pipelined ds loads)
    const unsigned short* bl = Bsh + (kt * 32 + lane) * 16;
    BF16x16 b[4];
#pragma unroll
    for (int j = 0; j < 4; ++j)
      b[j] = *(const BF16x16*)(bl + j * KT * 512);

#pragma unroll
    for (int j = 0; j < 4; ++j) {
      acc[0][j] = __builtin_amdgcn_wmma_f32_16x16x32_bf16(
          false, a0.v, false, b[j].v, (short)0, acc[0][j], false, false);
      acc[1][j] = __builtin_amdgcn_wmma_f32_16x16x32_bf16(
          false, a1.v, false, b[j].v, (short)0, acc[1][j], false, false);
    }
  }

  // C layout: VGPR r -> M = r (lanes 0..15) / 8+r (lanes 16..31); N = lane%16
  const int nloc = lane & 15;
  const int moff = (lane >= 16) ? 8 : 0;
#pragma unroll
  for (int mi = 0; mi < 2; ++mi) {
    const int mbase = (mt0 + mi) * 16 + moff;
#pragma unroll
    for (int j = 0; j < 4; ++j) {
      int ncol = (ntg * 4 + j) * 16 + nloc;
      float bv = bias ? bias[ncol] : 0.0f;
#pragma unroll
      for (int r = 0; r < 8; ++r)
        C[(long long)(mbase + r) * Nout + ncol] = acc[mi][j][r] + bv;
    }
  }
}

// ---------------- GRU gate combine (+ fused bf16 A-fragment pack) ----------
// h_new = (1-z)*tanh(gxn + r*ghn) + z*h ; also writes packed bf16 h_new.
__global__ void gru_combine(const float* __restrict__ gx, const float* __restrict__ gh,
                            const float* __restrict__ h, float* __restrict__ h_new,
                            unsigned short* __restrict__ hP, int total) {
  int idx = blockIdx.x * blockDim.x + threadIdx.x;
  if (idx >= total) return;
  int n = idx / HID;
  int j = idx - n * HID;
  long long base = (long long)n * (3 * HID);
  float r  = sigmoidf_(gx[base + j]           + gh[base + j]);
  float z  = sigmoidf_(gx[base + HID + j]     + gh[base + HID + j]);
  float nn = tanhf    (gx[base + 2 * HID + j] + r * gh[base + 2 * HID + j]);
  float v = (1.0f - z) * nn + z * h[idx];
  h_new[idx] = v;
  hP[frag_index(n, j, HID)] = f32_to_bf16(v);
}

// ---------------- GCN edge scatter ----------------
// agg[dst] += x[src] * norm[src], vectorized 4 floats per thread
__global__ void gcn_scatter(const float* __restrict__ X, const int* __restrict__ src,
                            const int* __restrict__ dst, const float* __restrict__ norm,
                            float* __restrict__ agg, int e, int F) {
  long long idx = (long long)blockIdx.x * blockDim.x + threadIdx.x;
  int fv = F >> 2;
  long long total = (long long)e * fv;
  if (idx >= total) return;
  int ei = (int)(idx / fv);
  int f  = (int)(idx - (long long)ei * fv) * 4;
  int s = src[ei], d = dst[ei];
  float ns = norm[s];
  const float4 xv = *(const float4*)(X + (long long)s * F + f);
  float* ag = agg + (long long)d * F + f;
  atomicAdd(ag + 0, xv.x * ns);
  atomicAdd(ag + 1, xv.y * ns);
  atomicAdd(ag + 2, xv.z * ns);
  atomicAdd(ag + 3, xv.w * ns);
}

// h1 = tanh(agg1 * norm[i] + b1), written directly as packed bf16 A-fragments
__global__ void act_norm_bias_tanh(const float* __restrict__ agg, const float* __restrict__ norm,
                                   const float* __restrict__ b, unsigned short* __restrict__ h1P,
                                   int total, int F) {
  int idx = blockIdx.x * blockDim.x + threadIdx.x;
  if (idx >= total) return;
  int n = idx / F;
  int f = idx - n * F;
  float v = tanhf(agg[idx] * norm[n] + b[f]);
  h1P[frag_index(n, f, F)] = f32_to_bf16(v);
}

// d = agg2*norm + b2 ; out[n,t,0:F]=temp ; out[n,t,F:2F]=d ; temp += dt*d (f32 + packed)
__global__ void finalize_step(const float* __restrict__ agg2, const float* __restrict__ norm,
                              const float* __restrict__ b2, const float* __restrict__ tvec,
                              float* __restrict__ temp, unsigned short* __restrict__ tempP,
                              float* __restrict__ out, int total, int tstep) {
  int idx = blockIdx.x * blockDim.x + threadIdx.x;
  if (idx >= total) return;
  int n = idx / IND;
  int f = idx - n * IND;
  float d  = agg2[idx] * norm[n] + b2[f];
  float tp = temp[idx];
  long long ob = ((long long)n * TT + tstep) * (2 * IND);
  out[ob + f]       = tp;   // outs[t] = carry temp (before update)
  out[ob + IND + f] = d;    // douts[t]
  float dt = (tstep < TT - 1) ? (tvec[tstep + 1] - tvec[tstep]) : 0.0f;
  float tn = tp + dt * d;
  temp[idx] = tn;
  tempP[frag_index(n, f, IND)] = f32_to_bf16(tn);
}

// ---------------- host helpers ----------------
static inline int cdiv(long long a, int b) { return (int)((a + b - 1) / b); }

static void launch_gemm(const unsigned short* Ap, const unsigned short* Bp, float* C,
                        const float* bias, int M, int K, int Nout, hipStream_t s) {
  dim3 grid(Nout / 64, (M / 32 + 7) / 8);
  wmma_gemm_packed<<<grid, 256, 0, s>>>(Ap, Bp, C, bias, M, K, Nout);
}

extern "C" void kernel_launch(void* const* d_in, const int* in_sizes, int n_in,
                              void* d_out, int out_size, void* d_ws, size_t ws_size,
                              hipStream_t stream) {
  (void)in_sizes; (void)n_in; (void)out_size; (void)ws_size;
  const float* x0   = (const float*)d_in[0];
  const float* tvec = (const float*)d_in[1];
  const int*   src  = (const int*)d_in[2];
  const int*   dst  = (const int*)d_in[3];
  const float* W_ih = (const float*)d_in[4];   // [3H, IN]
  const float* W_hh = (const float*)d_in[5];   // [3H, H]
  const float* b_ih = (const float*)d_in[6];
  const float* b_hh = (const float*)d_in[7];
  const float* W1   = (const float*)d_in[8];   // [H, GH]
  const float* b1   = (const float*)d_in[9];
  const float* W2   = (const float*)d_in[10];  // [GH, IN]
  const float* b2   = (const float*)d_in[11];
  float* out = (float*)d_out;

  // bump allocator over d_ws (256B aligned slabs)
  char* wp = (char*)d_ws;
  auto alloc = [&](size_t nbytes) -> void* {
    void* r = (void*)wp;
    wp += (nbytes + 255) & ~(size_t)255;
    return r;
  };
  float* temp  = (float*)alloc((size_t)NN * IND * 4);
  float* hbuf0 = (float*)alloc((size_t)NN * HID * 4);
  float* hbuf1 = (float*)alloc((size_t)NN * HID * 4);
  float* gx    = (float*)alloc((size_t)NN * 3 * HID * 4);
  float* gh    = (float*)alloc((size_t)NN * 3 * HID * 4);
  float* hw1   = (float*)alloc((size_t)NN * GH * 4);
  float* agg1  = (float*)alloc((size_t)NN * GH * 4);
  float* h1w2  = (float*)alloc((size_t)NN * IND * 4);
  float* agg2  = (float*)alloc((size_t)NN * IND * 4);
  float* deg   = (float*)alloc((size_t)NN * 4);
  float* norm  = (float*)alloc((size_t)NN * 4);
  unsigned short* tempP = (unsigned short*)alloc((size_t)NN * IND * 2);
  unsigned short* hP0   = (unsigned short*)alloc((size_t)NN * HID * 2);
  unsigned short* hP1   = (unsigned short*)alloc((size_t)NN * HID * 2);
  unsigned short* h1P   = (unsigned short*)alloc((size_t)NN * GH * 2);
  unsigned short* PWih  = (unsigned short*)alloc((size_t)3 * HID * IND * 2);
  unsigned short* PWhh  = (unsigned short*)alloc((size_t)3 * HID * HID * 2);
  unsigned short* PW1   = (unsigned short*)alloc((size_t)HID * GH * 2);
  unsigned short* PW2   = (unsigned short*)alloc((size_t)GH * IND * 2);

  const int BS = 256;

  // ---- one-time setup (deterministic, re-done every call) ----
  zero_f32<<<cdiv(NN, BS), BS, 0, stream>>>(deg, NN);
  deg_accum<<<cdiv(EE, BS), BS, 0, stream>>>(dst, deg, EE);
  norm_from_deg<<<cdiv(NN, BS), BS, 0, stream>>>(deg, norm, NN);

  pack_b<<<cdiv((long long)IND * 3 * HID, BS), BS, 0, stream>>>(W_ih, PWih, IND, 3 * HID, 1);
  pack_b<<<cdiv((long long)HID * 3 * HID, BS), BS, 0, stream>>>(W_hh, PWhh, HID, 3 * HID, 1);
  pack_b<<<cdiv((long long)HID * GH,      BS), BS, 0, stream>>>(W1,   PW1,  HID, GH,      0);
  pack_b<<<cdiv((long long)GH * IND,      BS), BS, 0, stream>>>(W2,   PW2,  GH,  IND,     0);

  init_temp<<<cdiv((long long)NN * IND, BS), BS, 0, stream>>>(x0, temp, tempP, NN * IND);
  zero_f32<<<cdiv((long long)NN * HID, BS), BS, 0, stream>>>(hbuf0, NN * HID);
  zero_f32<<<cdiv((long long)NN * HID / 2, BS), BS, 0, stream>>>((float*)hP0, NN * HID / 2);

  // ---- T rollout steps ----
  for (int t = 0; t < TT; ++t) {
    float* h  = (t & 1) ? hbuf1 : hbuf0;
    float* hn = (t & 1) ? hbuf0 : hbuf1;
    unsigned short* hPc = (t & 1) ? hP1 : hP0;
    unsigned short* hPn = (t & 1) ? hP0 : hP1;

    // GRU: gx = temp @ W_ih^T + b_ih ; gh = h @ W_hh^T + b_hh
    launch_gemm(tempP, PWih, gx, b_ih, NN, IND, 3 * HID, stream);
    launch_gemm(hPc,   PWhh, gh, b_hh, NN, HID, 3 * HID, stream);
    gru_combine<<<cdiv((long long)NN * HID, BS), BS, 0, stream>>>(gx, gh, h, hn, hPn, NN * HID);

    // GNN layer 1: agg(norm * (h_new @ W1)) -> tanh(+b1)
    launch_gemm(hPn, PW1, hw1, nullptr, NN, HID, GH, stream);
    zero_f32<<<cdiv((long long)NN * GH, BS), BS, 0, stream>>>(agg1, NN * GH);
    gcn_scatter<<<cdiv((long long)EE * (GH / 4), BS), BS, 0, stream>>>(hw1, src, dst, norm, agg1, EE, GH);
    act_norm_bias_tanh<<<cdiv((long long)NN * GH, BS), BS, 0, stream>>>(agg1, norm, b1, h1P, NN * GH, GH);

    // GNN layer 2: d = agg(norm * (h1 @ W2)) * norm + b2 ; outputs + Euler update
    launch_gemm(h1P, PW2, h1w2, nullptr, NN, GH, IND, stream);
    zero_f32<<<cdiv((long long)NN * IND, BS), BS, 0, stream>>>(agg2, NN * IND);
    gcn_scatter<<<cdiv((long long)EE * (IND / 4), BS), BS, 0, stream>>>(h1w2, src, dst, norm, agg2, EE, IND);
    finalize_step<<<cdiv((long long)NN * IND, BS), BS, 0, stream>>>(agg2, norm, b2, tvec, temp, tempP, out, NN * IND, t);
  }
}
